// MuserPLM_88648124991204
// MI455X (gfx1250) — matrix operand order repaired and checked by reading, not compiled, not run on hardware
//
#include <hip/hip_runtime.h>
#include <hip/hip_bf16.h>

// ---------------------------------------------------------------------------
// MI455X (gfx1250, wave32) Longformer-style encoder.
// GEMMs: f16 operands, TDM (tensor_load_to_lds) double-buffered LDS staging,
// v_wmma_f32_16x16x32_f16 compute. Banded attention: flash-style online
// softmax with WMMA QK^T and PV, TDM-staged Q tile.
// ---------------------------------------------------------------------------

typedef __attribute__((ext_vector_type(16))) _Float16 v16h;
typedef __attribute__((ext_vector_type(8)))  float    v8f;
typedef __attribute__((ext_vector_type(4)))  unsigned v4u;
typedef __attribute__((ext_vector_type(4)))  int      v4i;
typedef __attribute__((ext_vector_type(8)))  int      v8i;

#define Bc   2
#define Sc   4096
#define Dc   768
#define Hc   12
#define DHc  64
#define Wc   128
#define DFFc 3072
#define NEGV (-30000.0f)

// ---- Tensor Data Mover: 2D tile load global->LDS (CDNA5 ISA ch.8 D#) ------
// Descriptor: group0 = {count=1 | lds_addr | global_addr | type=2},
// group1 = {mask=0, data_size=1(2B), tensor_dim0/1, tile_dim0/1, dim0_stride}.
// tensor dims set equal to tile dims (tile fully in-range by construction).
__device__ __forceinline__ void tdm_load_2d(void* lds, const void* gsrc,
                                            unsigned tile_x, unsigned tile_y,
                                            unsigned stride_elems) {
  const unsigned lds_off = (unsigned)(size_t)lds;   // low 32 bits = LDS offset
  const unsigned long long ga = (unsigned long long)(size_t)gsrc;
  v4u g0;
  g0[0] = 1u;                                        // count=1, user D#
  g0[1] = lds_off;                                   // bits 63:32 lds_addr
  g0[2] = (unsigned)(ga & 0xffffffffull);            // global_addr[31:0]
  g0[3] = (unsigned)((ga >> 32) & 0x01ffffffull) | 0x80000000u; // addr[56:32]|type=2
  v8i g1;
  g1[0] = (int)(1u << 16);                           // data_size=1 (2 bytes)
  g1[1] = (int)((tile_x & 0xffffu) << 16);           // tensor_dim0[15:0]
  g1[2] = (int)((tile_x >> 16) | ((tile_y & 0xffffu) << 16)); // dim0 hi | dim1 lo
  g1[3] = (int)((tile_y >> 16) | (tile_x << 16));    // dim1 hi | tile_dim0
  g1[4] = (int)(tile_y & 0xffffu);                   // tile_dim1 (tile_dim2=0)
  g1[5] = (int)stride_elems;                         // tensor_dim0_stride[31:0]
  g1[6] = 0;
  g1[7] = 0;
  const v4i z4 = {0, 0, 0, 0};
#if __clang_major__ >= 23
  const v8i z8 = {0, 0, 0, 0, 0, 0, 0, 0};
  __builtin_amdgcn_tensor_load_to_lds(g0, g1, z4, z4, z8, 0);
#else
  __builtin_amdgcn_tensor_load_to_lds(g0, g1, z4, z4, 0);
#endif
}

// ---- WMMA fragment loaders (CDNA5 ISA 7.12.2 layouts, wave32) -------------
__device__ __forceinline__ v16h frag_a(const _Float16* base, int ld, int koff) {
  const int lane = threadIdx.x & 31;
  const int m = lane & 15, hf = lane >> 4;
  const _Float16* p = base + m * ld + koff + hf * 8;
  v16h f;
#pragma unroll
  for (int r = 0; r < 8; ++r) {
    const int k0 = ((r & 3) << 1) + ((r >> 2) << 4);
    f[2 * r]     = p[k0];
    f[2 * r + 1] = p[k0 + 1];
  }
  return f;
}
__device__ __forceinline__ v16h frag_b(const _Float16* base, int ld, int koff) {
  const int lane = threadIdx.x & 31;
  const int n = lane & 15, hf = lane >> 4;
  const _Float16* p = base + n * ld + koff + hf * 8;
  v16h f;
#pragma unroll
  for (int r = 0; r < 8; ++r) {
    const int k0 = ((r & 3) << 1) + ((r >> 2) << 4);
    f[2 * r]     = p[k0];
    f[2 * r + 1] = p[k0 + 1];
  }
  return f;
}

__device__ __forceinline__ float gelu_f(float x) {
  const float c = 0.7978845608028654f;
  return 0.5f * x * (1.0f + tanhf(c * (x + 0.044715f * x * x * x)));
}

// ---------------------------------------------------------------------------
// Elementwise f32 -> f16 convert (n multiple of 4)
// ---------------------------------------------------------------------------
__global__ __launch_bounds__(256) void cvt_f16_kernel(
    const float* __restrict__ src, _Float16* __restrict__ dst, int n) {
  const int i = (blockIdx.x * 256 + threadIdx.x) * 4;
  if (i + 3 < n) {
    const float4 v = *(const float4*)(src + i);
    dst[i]     = (_Float16)v.x;
    dst[i + 1] = (_Float16)v.y;
    dst[i + 2] = (_Float16)v.z;
    dst[i + 3] = (_Float16)v.w;
  }
}

// ---------------------------------------------------------------------------
// Transpose-convert weights: W[K][N] f32  ->  Bt[N][K] f16 (64x64 tiles)
// ---------------------------------------------------------------------------
__global__ __launch_bounds__(256) void cvt_tr_kernel(
    const float* __restrict__ W, _Float16* __restrict__ Bt, int K, int N) {
  __shared__ float t[64][65];
  const int tid = threadIdx.x;
  const int kb = blockIdx.x * 64;
  const int nb = blockIdx.y * 64;
  const int c = tid & 63;
  const int r4 = tid >> 6;  // 0..3
#pragma unroll
  for (int i = 0; i < 16; ++i) {
    const int kk = r4 + i * 4;
    t[kk][c] = W[(size_t)(kb + kk) * N + nb + c];
  }
  __syncthreads();
#pragma unroll
  for (int i = 0; i < 16; ++i) {
    const int nn = r4 + i * 4;
    Bt[(size_t)(nb + nn) * K + kb + c] = (_Float16)t[c][nn];
  }
}

// ---------------------------------------------------------------------------
// Embedding gather + layernorm
// ---------------------------------------------------------------------------
__global__ __launch_bounds__(256) void embed_ln_kernel(
    const int* __restrict__ inputx, const float* __restrict__ tok_emb,
    const float* __restrict__ pos_emb, const float* __restrict__ g,
    const float* __restrict__ bb, float* __restrict__ x) {
  const int row = blockIdx.x;
  const int s = row & (Sc - 1);
  const int tid = threadIdx.x;
  __shared__ float vals[Dc];
  __shared__ float red[256];
  const int tok = inputx[row];
  float local = 0.f;
  for (int i = tid; i < Dc; i += 256) {
    float v = tok_emb[(size_t)tok * Dc + i] + pos_emb[(size_t)s * Dc + i];
    vals[i] = v;
    local += v;
  }
  red[tid] = local;
  __syncthreads();
  for (int st = 128; st > 0; st >>= 1) {
    if (tid < st) red[tid] += red[tid + st];
    __syncthreads();
  }
  const float mu = red[0] / (float)Dc;
  __syncthreads();
  float lv = 0.f;
  for (int i = tid; i < Dc; i += 256) { float d = vals[i] - mu; lv += d * d; }
  red[tid] = lv;
  __syncthreads();
  for (int st = 128; st > 0; st >>= 1) {
    if (tid < st) red[tid] += red[tid + st];
    __syncthreads();
  }
  const float rstd = rsqrtf(red[0] / (float)Dc + 1e-5f);
  for (int i = tid; i < Dc; i += 256)
    x[(size_t)row * Dc + i] = (vals[i] - mu) * rstd * g[i] + bb[i];
}

// ---------------------------------------------------------------------------
// Residual + layernorm (in-place safe)
// ---------------------------------------------------------------------------
__global__ __launch_bounds__(256) void add_ln_kernel(
    const float* __restrict__ xin, const float* __restrict__ y,
    const float* __restrict__ g, const float* __restrict__ bb,
    float* __restrict__ xout) {
  const int row = blockIdx.x;
  const int tid = threadIdx.x;
  __shared__ float vals[Dc];
  __shared__ float red[256];
  float local = 0.f;
  for (int i = tid; i < Dc; i += 256) {
    float v = xin[(size_t)row * Dc + i] + y[(size_t)row * Dc + i];
    vals[i] = v;
    local += v;
  }
  red[tid] = local;
  __syncthreads();
  for (int st = 128; st > 0; st >>= 1) {
    if (tid < st) red[tid] += red[tid + st];
    __syncthreads();
  }
  const float mu = red[0] / (float)Dc;
  __syncthreads();
  float lv = 0.f;
  for (int i = tid; i < Dc; i += 256) { float d = vals[i] - mu; lv += d * d; }
  red[tid] = lv;
  __syncthreads();
  for (int st = 128; st > 0; st >>= 1) {
    if (tid < st) red[tid] += red[tid + st];
    __syncthreads();
  }
  const float rstd = rsqrtf(red[0] / (float)Dc + 1e-5f);
  for (int i = tid; i < Dc; i += 256)
    xout[(size_t)row * Dc + i] = (vals[i] - mu) * rstd * g[i] + bb[i];
}

// ---------------------------------------------------------------------------
// TDM + WMMA GEMM:  C = act(A[M,K] * Bt[N,K]^T + bias)
// A f16 row-major, Bt f16 n-major (pre-transposed weights). 128x64 block tile,
// double-buffered LDS, TDM-staged, one wave = one 16-row stripe.
// Cf (f32) and/or Ch (f16) outputs.
// ---------------------------------------------------------------------------
__global__ __launch_bounds__(256) void gemm_tdm_kernel(
    const _Float16* __restrict__ A, const _Float16* __restrict__ Bt,
    const float* __restrict__ bias, float* __restrict__ Cf,
    _Float16* __restrict__ Ch, int M, int N, int K, int act) {
  __shared__ _Float16 As[2][128][32];
  __shared__ _Float16 Bs[2][64][32];
  const int tid = threadIdx.x;
  const int wave = tid >> 5;
  const int lane = tid & 31;
  const int colL = lane & 15;
  const int halfL = lane >> 4;
  const int m0 = blockIdx.x * 128;
  const int n0 = blockIdx.y * 64;

  const v8f zero8 = {0.f, 0.f, 0.f, 0.f, 0.f, 0.f, 0.f, 0.f};
  v8f acc[4] = {zero8, zero8, zero8, zero8};

  const int KT = K / 32;
  if (tid < 32) {  // wave 0 drives the TDM
    tdm_load_2d(&As[0][0][0], A + (size_t)m0 * K, 32, 128, (unsigned)K);
    tdm_load_2d(&Bs[0][0][0], Bt + (size_t)n0 * K, 32, 64, (unsigned)K);
  }
  int buf = 0;
  for (int kt = 0; kt < KT; ++kt) {
    if (tid < 32) {
      if (kt + 1 < KT) {  // prefetch next tile pair into the other buffer
        const size_t k1 = (size_t)(kt + 1) * 32;
        tdm_load_2d(&As[buf ^ 1][0][0], A + (size_t)m0 * K + k1, 32, 128, (unsigned)K);
        tdm_load_2d(&Bs[buf ^ 1][0][0], Bt + (size_t)n0 * K + k1, 32, 64, (unsigned)K);
        __builtin_amdgcn_s_wait_tensorcnt(2);  // current pair done, next in flight
      } else {
        __builtin_amdgcn_s_wait_tensorcnt(0);
      }
    }
    __syncthreads();
    const v16h a = frag_a(&As[buf][wave * 16][0], 32, 0);
#pragma unroll
    for (int t = 0; t < 4; ++t) {
      const v16h b = frag_b(&Bs[buf][t * 16][0], 32, 0);
      acc[t] = __builtin_amdgcn_wmma_f32_16x16x32_f16(
          false, a, false, b, (short)0, acc[t], false, false);
    }
    __syncthreads();  // everyone done with buf before TDM refills it
    buf ^= 1;
  }
#pragma unroll
  for (int t = 0; t < 4; ++t) {
    const int n = n0 + t * 16 + colL;
    const float bv = bias[n];
#pragma unroll
    for (int r = 0; r < 8; ++r) {
      const int m = m0 + wave * 16 + r + 8 * halfL;
      float v = acc[t][r] + bv;
      if (act == 1) v = gelu_f(v);
      if (Cf) Cf[(size_t)m * N + n] = v;
      if (Ch) Ch[(size_t)m * N + n] = (_Float16)v;
    }
  }
}

// ---------------------------------------------------------------------------
// Banded local attention (flash-style online softmax over 3W band + G=1 col).
// f16 Q/K/V inputs; Q tile staged by TDM. Grid: (nb, H, B), 8 waves/block.
// ---------------------------------------------------------------------------
__global__ __launch_bounds__(256) void banded_attn_kernel(
    const _Float16* __restrict__ qb, const _Float16* __restrict__ kb,
    const _Float16* __restrict__ vb, const _Float16* __restrict__ kgb,
    const _Float16* __restrict__ vgb, const float* __restrict__ mask,
    float* __restrict__ attn) {
  const int nblk = blockIdx.x;
  const int h = blockIdx.y;
  const int b = blockIdx.z;
  const int tid = threadIdx.x;
  const int wave = tid >> 5;
  const int lane = tid & 31;
  const int colL = lane & 15;
  const int halfL = lane >> 4;
  const float scale = 0.125f;  // 1/sqrt(64)

  __shared__ _Float16 Qs[128][64];     // TDM-staged
  __shared__ _Float16 Kt[32][64];      // key-major
  __shared__ _Float16 Vts[64][32];     // dh-major (transposed V)
  __shared__ float Ms[32];
  __shared__ _Float16 Pst[8][16][32];  // per-wave P stash (D->A relayout)

  const size_t hoff = (size_t)h * DHc;
  const int s0base = nblk * Wc;

  if (tid < 32) {
    tdm_load_2d(&Qs[0][0], qb + ((size_t)(b * Sc + s0base)) * Dc + hoff,
                64, 128, (unsigned)Dc);
    __builtin_amdgcn_s_wait_tensorcnt(0);
  }
  __syncthreads();

  const v16h aqlo = frag_a(&Qs[wave * 16][0], 64, 0);
  const v16h aqhi = frag_a(&Qs[wave * 16][0], 64, 32);

  // init running stats with the single global column (key position 0)
  float mrun[8], lrun[8];
  v8f oacc[4];
  const _Float16* kg0 = kgb + ((size_t)(b * Sc)) * Dc + hoff;
  const _Float16* vg0 = vgb + ((size_t)(b * Sc)) * Dc + hoff;
  const float gmask = mask[b * Sc];
#pragma unroll
  for (int r = 0; r < 8; ++r) {
    const int row = wave * 16 + r + 8 * halfL;
    float d = 0.f;
    const int d0 = colL * 4;
#pragma unroll
    for (int i = 0; i < 4; ++i) d += (float)Qs[row][d0 + i] * (float)kg0[d0 + i];
    for (int sft = 1; sft < 16; sft <<= 1) d += __shfl_xor(d, sft, 16);
    d *= scale;
    d = (gmask > 0.f) ? d : NEGV;
    mrun[r] = d;
    lrun[r] = 1.f;
  }
#pragma unroll
  for (int nt = 0; nt < 4; ++nt) {
    const float vv = (float)vg0[nt * 16 + colL];
#pragma unroll
    for (int r = 0; r < 8; ++r) oacc[nt][r] = vv;  // exp(scg - m) == 1
  }

  const v8f zero8 = {0.f, 0.f, 0.f, 0.f, 0.f, 0.f, 0.f, 0.f};

  for (int tp = 0; tp < 12; ++tp) {
    __syncthreads();
    {
      const int kk = tid >> 3;
      const int d0 = (tid & 7) * 8;
      const int kpos = (nblk - 1) * Wc + tp * 32 + kk;
      if (kpos >= 0 && kpos < Sc) {
        const _Float16* ksrc = kb + ((size_t)(b * Sc + kpos)) * Dc + hoff + d0;
        const _Float16* vsrc = vb + ((size_t)(b * Sc + kpos)) * Dc + hoff + d0;
        *(uint4*)(&Kt[kk][d0]) = *(const uint4*)ksrc;
        alignas(16) _Float16 vtmp[8];
        *(uint4*)vtmp = *(const uint4*)vsrc;
#pragma unroll
        for (int i = 0; i < 8; ++i) Vts[d0 + i][kk] = vtmp[i];
      } else {
        const uint4 z = {0u, 0u, 0u, 0u};
        *(uint4*)(&Kt[kk][d0]) = z;
#pragma unroll
        for (int i = 0; i < 8; ++i) Vts[d0 + i][kk] = (_Float16)0.f;
      }
      if (tid < 32) {
        const int kp2 = (nblk - 1) * Wc + tp * 32 + tid;
        Ms[tid] = (kp2 >= 0 && kp2 < Sc) ? mask[b * Sc + kp2] : 0.f;
      }
    }
    __syncthreads();

    const v16h bk0lo = frag_b(&Kt[0][0], 64, 0);
    const v16h bk0hi = frag_b(&Kt[0][0], 64, 32);
    const v16h bk1lo = frag_b(&Kt[16][0], 64, 0);
    const v16h bk1hi = frag_b(&Kt[16][0], 64, 32);
    v8f s0 = __builtin_amdgcn_wmma_f32_16x16x32_f16(false, aqlo, false, bk0lo,
                                                    (short)0, zero8, false, false);
    s0 = __builtin_amdgcn_wmma_f32_16x16x32_f16(false, aqhi, false, bk0hi,
                                                (short)0, s0, false, false);
    v8f s1 = __builtin_amdgcn_wmma_f32_16x16x32_f16(false, aqlo, false, bk1lo,
                                                    (short)0, zero8, false, false);
    s1 = __builtin_amdgcn_wmma_f32_16x16x32_f16(false, aqhi, false, bk1hi,
                                                (short)0, s1, false, false);

#pragma unroll
    for (int r = 0; r < 8; ++r) {
      const int qi = wave * 16 + r + 8 * halfL;
      const int j0 = tp * 32 + colL;
      const int j1 = j0 + 16;
      const int kp0 = (nblk - 1) * Wc + j0;
      const int kp1 = kp0 + 16;
      float v0 = s0[r] * scale;
      float v1 = s1[r] * scale;
      const bool ok0 = (j0 >= qi) && (j0 <= qi + 2 * Wc) && (kp0 >= 1) &&
                       (kp0 < Sc) && (Ms[colL] > 0.f);
      const bool ok1 = (j1 >= qi) && (j1 <= qi + 2 * Wc) && (kp1 >= 1) &&
                       (kp1 < Sc) && (Ms[colL + 16] > 0.f);
      v0 = ok0 ? v0 : NEGV;
      v1 = ok1 ? v1 : NEGV;
      float mt = fmaxf(v0, v1);
      for (int sft = 1; sft < 16; sft <<= 1) mt = fmaxf(mt, __shfl_xor(mt, sft, 16));
      const float newm = fmaxf(mrun[r], mt);
      const float p0 = __expf(v0 - newm);
      const float p1 = __expf(v1 - newm);
      float rs = p0 + p1;
      for (int sft = 1; sft < 16; sft <<= 1) rs += __shfl_xor(rs, sft, 16);
      const float alpha = __expf(mrun[r] - newm);
      lrun[r] = lrun[r] * alpha + rs;
      mrun[r] = newm;
#pragma unroll
      for (int nt = 0; nt < 4; ++nt) oacc[nt][r] *= alpha;
      const int rloc = r + 8 * halfL;
      Pst[wave][rloc][colL] = (_Float16)p0;
      Pst[wave][rloc][16 + colL] = (_Float16)p1;
    }

    const v16h pa = frag_a(&Pst[wave][0][0], 32, 0);
#pragma unroll
    for (int nt = 0; nt < 4; ++nt) {
      const v16h bv = frag_b(&Vts[nt * 16][0], 32, 0);
      oacc[nt] = __builtin_amdgcn_wmma_f32_16x16x32_f16(
          false, pa, false, bv, (short)0, oacc[nt], false, false);
    }
  }

#pragma unroll
  for (int nt = 0; nt < 4; ++nt) {
#pragma unroll
    for (int r = 0; r < 8; ++r) {
      const int row = wave * 16 + r + 8 * halfL;
      attn[((size_t)(b * Sc + s0base + row)) * Dc + hoff + nt * 16 + colL] =
          oacc[nt][r] / lrun[r];
    }
  }
}

// ---------------------------------------------------------------------------
// Global-token attention (row s=0 attends to all S). Grid: (H, B).
// ---------------------------------------------------------------------------
__global__ __launch_bounds__(256) void global_attn_kernel(
    const _Float16* __restrict__ qgb, const _Float16* __restrict__ kgb,
    const _Float16* __restrict__ vgb, const float* __restrict__ mask,
    float* __restrict__ attn) {
  const int h = blockIdx.x;
  const int b = blockIdx.y;
  const int tid = threadIdx.x;
  __shared__ float qg0[DHc];
  __shared__ float scs[Sc];
  __shared__ float red[256];
  __shared__ float part[4][DHc];
  const size_t hoff = (size_t)h * DHc;
  if (tid < DHc)
    qg0[tid] = (float)qgb[((size_t)(b * Sc)) * Dc + hoff + tid] * 0.125f;
  __syncthreads();
  float lmax = -1e30f;
  for (int s = tid; s < Sc; s += 256) {
    const _Float16* kr = kgb + ((size_t)(b * Sc + s)) * Dc + hoff;
    float d = 0.f;
#pragma unroll 8
    for (int i = 0; i < DHc; ++i) d += qg0[i] * (float)kr[i];
    d = (mask[b * Sc + s] > 0.f) ? d : NEGV;
    scs[s] = d;
    lmax = fmaxf(lmax, d);
  }
  red[tid] = lmax;
  __syncthreads();
  for (int st = 128; st > 0; st >>= 1) {
    if (tid < st) red[tid] = fmaxf(red[tid], red[tid + st]);
    __syncthreads();
  }
  const float m = red[0];
  __syncthreads();
  float ls = 0.f;
  for (int s = tid; s < Sc; s += 256) {
    const float p = __expf(scs[s] - m);
    scs[s] = p;
    ls += p;
  }
  red[tid] = ls;
  __syncthreads();
  for (int st = 128; st > 0; st >>= 1) {
    if (tid < st) red[tid] += red[tid + st];
    __syncthreads();
  }
  const float lsum = red[0];
  __syncthreads();
  const int dh = tid & 63;
  const int ch = tid >> 6;
  float acc = 0.f;
  for (int s = ch * (Sc / 4); s < (ch + 1) * (Sc / 4); ++s)
    acc += scs[s] * (float)vgb[((size_t)(b * Sc + s)) * Dc + hoff + dh];
  part[ch][dh] = acc;
  __syncthreads();
  if (tid < DHc) {
    const float o = (part[0][tid] + part[1][tid] + part[2][tid] + part[3][tid]) / lsum;
    attn[((size_t)(b * Sc)) * Dc + hoff + tid] = o;  // overwrite s = 0
  }
}

// ---------------------------------------------------------------------------
// Pooled head: pooled = tanh(x[:,0] @ Wp + bp); logits; mean NLL. One block.
// ---------------------------------------------------------------------------
__global__ __launch_bounds__(256) void pool_loss_kernel(
    const float* __restrict__ x, const float* __restrict__ Wp,
    const float* __restrict__ bp, const float* __restrict__ Wfc,
    const float* __restrict__ bfc, const int* __restrict__ labels,
    float* __restrict__ out) {
  __shared__ float pooled[Bc][Dc];
  __shared__ float logit[Bc][2];
  const int tid = threadIdx.x;
  for (int b = 0; b < Bc; ++b) {
    const float* xr = x + ((size_t)(b * Sc)) * Dc;
    for (int j = tid; j < Dc; j += 256) {
      float d = bp[j];
      for (int i = 0; i < Dc; ++i) d += xr[i] * Wp[(size_t)i * Dc + j];
      pooled[b][j] = tanhf(d);
    }
  }
  __syncthreads();
  if (tid < 4) {
    const int b = tid >> 1, c = tid & 1;
    float d = bfc[c];
    for (int j = 0; j < Dc; ++j) d += pooled[b][j] * Wfc[j * 2 + c];
    logit[b][c] = d;
  }
  __syncthreads();
  if (tid == 0) {
    float loss = 0.f;
    for (int b = 0; b < Bc; ++b) {
      const float m = fmaxf(logit[b][0], logit[b][1]);
      const float z = __expf(logit[b][0] - m) + __expf(logit[b][1] - m);
      loss -= (logit[b][labels[b]] - m - logf(z));
    }
    out[0] = loss / (float)Bc;
  }
}

// ---------------------------------------------------------------------------
// Orchestration. Workspace: 4 f32 slots + f16 region (activations,
// projections, transposed f16 weights). FFN hidden aliases freed proj slots.
// ---------------------------------------------------------------------------
extern "C" void kernel_launch(void* const* d_in, const int* in_sizes, int n_in,
                              void* d_out, int out_size, void* d_ws, size_t ws_size,
                              hipStream_t stream) {
  const int*   inputx   = (const int*)d_in[0];
  const float* mask     = (const float*)d_in[1];
  const int*   labels   = (const int*)d_in[2];
  const float* tok_emb  = (const float*)d_in[3];
  const float* pos_emb  = (const float*)d_in[4];
  const float* ln_emb_g = (const float*)d_in[5];
  const float* ln_emb_b = (const float*)d_in[6];
  const float* Wq  = (const float*)d_in[7];
  const float* bq  = (const float*)d_in[8];
  const float* Wk  = (const float*)d_in[9];
  const float* bk  = (const float*)d_in[10];
  const float* Wv  = (const float*)d_in[11];
  const float* bv  = (const float*)d_in[12];
  const float* Wqg = (const float*)d_in[13];
  const float* bqg = (const float*)d_in[14];
  const float* Wkg = (const float*)d_in[15];
  const float* bkg = (const float*)d_in[16];
  const float* Wvg = (const float*)d_in[17];
  const float* bvg = (const float*)d_in[18];
  const float* Wo  = (const float*)d_in[19];
  const float* bo  = (const float*)d_in[20];
  const float* ln1_g = (const float*)d_in[21];
  const float* ln1_b = (const float*)d_in[22];
  const float* W1  = (const float*)d_in[23];
  const float* b1  = (const float*)d_in[24];
  const float* W2  = (const float*)d_in[25];
  const float* b2  = (const float*)d_in[26];
  const float* ln2_g = (const float*)d_in[27];
  const float* ln2_b = (const float*)d_in[28];
  const float* Wp  = (const float*)d_in[29];
  const float* bp  = (const float*)d_in[30];
  const float* Wfc = (const float*)d_in[31];
  const float* bfc = (const float*)d_in[32];

  const size_t RC = (size_t)Bc * Sc * Dc;  // 8192*768
  const size_t WSZ = (size_t)Dc * Dc;      // 589824
  float* ws_f = (float*)d_ws;
  float* x    = ws_f + 0 * RC;
  float* attn = ws_f + 1 * RC;
  float* tmp  = ws_f + 2 * RC;
  float* tmp2 = ws_f + 3 * RC;
  _Float16* hb     = (_Float16*)(ws_f + 4 * RC);
  _Float16* x_h    = hb + 0 * RC;
  _Float16* attn_h = hb + 1 * RC;
  _Float16* qh     = hb + 2 * RC;
  _Float16* kh     = hb + 3 * RC;
  _Float16* vh     = hb + 4 * RC;
  _Float16* qgh    = hb + 5 * RC;
  _Float16* kgh    = hb + 6 * RC;
  _Float16* vgh    = hb + 7 * RC;
  _Float16* h1_h   = qh;  // 8192x3072 halfs = 4*RC, aliases qh..qgh (free then)
  _Float16* wT     = hb + 8 * RC;       // 7 x [768][768]
  _Float16* w1T    = wT + 7 * WSZ;      // [3072][768]
  _Float16* w2T    = w1T + (size_t)Dc * DFFc;  // [768][3072]

  const dim3 blk(256);
  const int Mrows = Bc * Sc;  // 8192

  embed_ln_kernel<<<dim3(Mrows), blk, 0, stream>>>(inputx, tok_emb, pos_emb,
                                                   ln_emb_g, ln_emb_b, x);

  for (int l = 0; l < 2; ++l) {
    const size_t wo = (size_t)l * WSZ;
    const size_t bo1 = (size_t)l * Dc;
    // --- transpose-convert this layer's weights to f16 [N][K] ---
    const float* Wlist[7] = {Wq + wo, Wk + wo, Wv + wo, Wqg + wo,
                             Wkg + wo, Wvg + wo, Wo + wo};
    for (int i = 0; i < 7; ++i)
      cvt_tr_kernel<<<dim3(Dc / 64, Dc / 64), blk, 0, stream>>>(
          Wlist[i], wT + (size_t)i * WSZ, Dc, Dc);
    cvt_tr_kernel<<<dim3(Dc / 64, DFFc / 64), blk, 0, stream>>>(
        W1 + (size_t)l * Dc * DFFc, w1T, Dc, DFFc);
    cvt_tr_kernel<<<dim3(DFFc / 64, Dc / 64), blk, 0, stream>>>(
        W2 + (size_t)l * DFFc * Dc, w2T, DFFc, Dc);

    // --- projections (f16 in, f16 out) ---
    cvt_f16_kernel<<<dim3((int)(RC / 1024)), blk, 0, stream>>>(x, x_h, (int)RC);
    const dim3 gproj(Mrows / 128, Dc / 64);
    _Float16* Pout[6] = {qh, kh, vh, qgh, kgh, vgh};
    const float* Blist[6] = {bq + bo1, bk + bo1, bv + bo1,
                             bqg + bo1, bkg + bo1, bvg + bo1};
    for (int i = 0; i < 6; ++i)
      gemm_tdm_kernel<<<gproj, blk, 0, stream>>>(
          x_h, wT + (size_t)i * WSZ, Blist[i], (float*)nullptr, Pout[i],
          Mrows, Dc, Dc, 0);

    // --- attention ---
    banded_attn_kernel<<<dim3(Sc / Wc, Hc, Bc), blk, 0, stream>>>(
        qh, kh, vh, kgh, vgh, mask, attn);
    global_attn_kernel<<<dim3(Hc, Bc), blk, 0, stream>>>(qgh, kgh, vgh, mask, attn);

    // --- output projection + residual LN ---
    cvt_f16_kernel<<<dim3((int)(RC / 1024)), blk, 0, stream>>>(attn, attn_h, (int)RC);
    gemm_tdm_kernel<<<gproj, blk, 0, stream>>>(
        attn_h, wT + 6 * WSZ, bo + bo1, tmp, (_Float16*)nullptr,
        Mrows, Dc, Dc, 0);
    add_ln_kernel<<<dim3(Mrows), blk, 0, stream>>>(x, tmp, ln1_g + bo1,
                                                   ln1_b + bo1, x);

    // --- FFN ---
    cvt_f16_kernel<<<dim3((int)(RC / 1024)), blk, 0, stream>>>(x, x_h, (int)RC);
    gemm_tdm_kernel<<<dim3(Mrows / 128, DFFc / 64), blk, 0, stream>>>(
        x_h, w1T, b1 + (size_t)l * DFFc, (float*)nullptr, h1_h,
        Mrows, DFFc, Dc, 1);
    gemm_tdm_kernel<<<dim3(Mrows / 128, Dc / 64), blk, 0, stream>>>(
        h1_h, w2T, b2 + bo1, tmp2, (_Float16*)nullptr, Mrows, Dc, DFFc, 0);
    add_ln_kernel<<<dim3(Mrows), blk, 0, stream>>>(x, tmp2, ln2_g + bo1,
                                                   ln2_b + bo1, x);
  }

  pool_loss_kernel<<<dim3(1), blk, 0, stream>>>(x, Wp, bp, Wfc, bfc, labels,
                                                (float*)d_out);
}
